// EfficientSparseGCN_81217831568032
// MI455X (gfx1250) — matrix operand order repaired and checked by reading, not compiled, not run on hardware
//
#include <hip/hip_runtime.h>
#include <hip/hip_bf16.h>

typedef __attribute__((ext_vector_type(16))) __bf16       v16bf;
typedef __attribute__((ext_vector_type(8)))  float        v8f;
typedef __attribute__((ext_vector_type(4)))  float        v4f;
typedef __attribute__((ext_vector_type(4)))  unsigned int v4u;
typedef __attribute__((ext_vector_type(8)))  int          v8i;
typedef __attribute__((ext_vector_type(4)))  int          v4i;

#define LN_EPS    1e-5f
#define WT_STRIDE 136   // 128 + 8 bf16 pad (= 64 dw + 4 dw) -> no LDS bank conflicts
#define TILES_PER_BLOCK 8

static __device__ __forceinline__ __bf16 to_bf16(float f) { return (__bf16)f; }

// ---------------- zero the SpMM accumulator ----------------
__global__ void gcn_zero_kernel(float4* __restrict__ p, int n4) {
    int i = blockIdx.x * blockDim.x + threadIdx.x;
    if (i < n4) p[i] = make_float4(0.f, 0.f, 0.f, 0.f);
}

// ---------------- one-time W^T -> bf16 conversion (n-major rows of 128) ----------------
__global__ void gcn_wt_prep_kernel(const float* __restrict__ W, unsigned short* __restrict__ Wt) {
    int i = blockIdx.x * 256 + threadIdx.x;
    if (i < 128 * 128) {
        const int n = i >> 7, k = i & 127;
        Wt[i] = __builtin_bit_cast(unsigned short, to_bf16(W[k * 128 + n]));
    }
}

// ---------------- edge scatter: y[b,row,:] += val * x[b,col,:] ----------------
// one wave32 per edge; 32 lanes x 8 elems = 256 floats (B*D); L2-resident atomics
__global__ void __launch_bounds__(256)
gcn_edge_spmm_kernel(const float* __restrict__ x,
                     const int*   __restrict__ rows,
                     const int*   __restrict__ cols,
                     const float* __restrict__ vals,
                     float*       __restrict__ y,
                     int E, int N) {
    const int wave = threadIdx.x >> 5;
    const int lane = threadIdx.x & 31;
    const int e = blockIdx.x * 8 + wave;
    if (e >= E) return;
    const int   r = rows[e];
    const int   c = cols[e];
    const float v = vals[e];
    const size_t ND = (size_t)N * 128;
#pragma unroll
    for (int b = 0; b < 2; ++b) {
        const float* xs = x + b * ND + (size_t)c * 128;
        float*       yd = y + b * ND + (size_t)r * 128;
#pragma unroll
        for (int i = 0; i < 4; ++i) {
            const int d = lane + i * 32;
            unsafeAtomicAdd(&yd[d], v * xs[d]);   // global_atomic_add_f32
        }
    }
}

// ---------------- fused GEMM (bf16 WMMA) + bias + LayerNorm + GELU ----------------
// block = 256 threads = 8 waves; W^T staged into LDS once per block via TDM
// (tensor_load_to_lds, hardware row padding), then 8 row-tiles of 16 rows each.
__global__ void __launch_bounds__(256)
gcn_gemm_ln_gelu_kernel(const float*          __restrict__ y,
                        const unsigned short* __restrict__ Wt,   // [128][128] bf16, n-major
                        const float*          __restrict__ bias,
                        const float*          __restrict__ gamma,
                        const float*          __restrict__ beta,
                        float*                __restrict__ out,
                        int Mtot) {
    __shared__ __align__(16) __bf16 sW[128 * WT_STRIDE];  // W^T bf16, padded rows
    __shared__ float sH[16 * 128];
    __shared__ float sSum[256];
    __shared__ float sSq[256];
    __shared__ float sMu[16];
    __shared__ float sRs[16];

    const int tid = threadIdx.x;

    // ---- TDM: DMA 128 rows x 256B of bf16 W^T into LDS, +16B pad per row ----
    if (tid == 0) {
        const unsigned long long ga = (unsigned long long)(uintptr_t)Wt;
        const unsigned lds = (unsigned)(uintptr_t)&sW[0];   // flat->LDS: low 32 bits
        // D# group 0: count=1 | lds_addr | global_addr[56:0] | type=2
        v4u g0 = { 1u, lds, (unsigned)ga, (unsigned)(ga >> 32) | (2u << 30) };
        // D# group 1: data_size=4B, pad_enable, pad_interval=64dw(code 5), pad_amount=4dw(code 3)
        //             tensor_dim0=64dw, tensor_dim1=128, tile_dim0=64dw, tile_dim1=128,
        //             tensor_dim0_stride=64dw
        v8i g1 = { (int)((2u << 16) | (1u << 20) | (5u << 22) | (3u << 25)),
                   (int)(64u << 16),    // [63:48]  tensor_dim0[15:0]
                   (int)(128u << 16),   // [95:80]  tensor_dim1[15:0]
                   (int)(64u << 16),    // [127:112] tile_dim0
                   (int)128,            // [143:128] tile_dim1
                   (int)64,             // [191:160] tensor_dim0_stride
                   0, 0 };
        v4i gz = { 0, 0, 0, 0 };
#if __has_include(<hip/amd_detail/amd_gfx1250_TDM.h>)
        v8i gz8 = { 0, 0, 0, 0, 0, 0, 0, 0 };
        __builtin_amdgcn_tensor_load_to_lds(g0, g1, gz, gz, gz8, 0);   // 6-arg toolchain
#else
        __builtin_amdgcn_tensor_load_to_lds(g0, g1, gz, gz, 0);        // 5-arg ROCm 7.2
#endif
    }
    __builtin_amdgcn_s_wait_tensorcnt(0);
    __syncthreads();

    const int wave = tid >> 5;
    const int lane = tid & 31;
    const int hi   = lane >> 4;       // half-wave select (ISA A/B layouts)
    const int l16  = lane & 15;
    const int n0   = wave * 16;       // this wave's output-column tile
    const int col  = n0 + l16;
    const float bv = bias[col];

    // epilogue constants: thread t normalizes cols seg*8..seg*8+7 of row (t>>4)
    const int row = tid >> 4;
    const int seg = tid & 15;
    float gv[8], bev[8];
#pragma unroll
    for (int j = 0; j < 8; ++j) { gv[j] = gamma[seg * 8 + j]; bev[j] = beta[seg * 8 + j]; }

#pragma unroll 1
    for (int t = 0; t < TILES_PER_BLOCK; ++t) {
        const int rBase = (blockIdx.x * TILES_PER_BLOCK + t) * 16;
        if (rBase >= Mtot) break;                      // uniform across block

        int aRow = rBase + l16;
        if (aRow >= Mtot) aRow = Mtot - 1;             // clamp tail rows
        const float* yrow = y + (size_t)aRow * 128;

        v8f acc = {};
#pragma unroll
        for (int s = 0; s < 4; ++s) {
            const int k0 = s * 32;
            // A fragment: lane = row M=l16, K = k0 + hi*16 + 0..15 (streamed, NT)
            const float* ap = yrow + k0 + hi * 16;
            const v4f f0 = __builtin_nontemporal_load((const v4f*)(ap));
            const v4f f1 = __builtin_nontemporal_load((const v4f*)(ap + 4));
            const v4f f2 = __builtin_nontemporal_load((const v4f*)(ap + 8));
            const v4f f3 = __builtin_nontemporal_load((const v4f*)(ap + 12));
            v16bf a;
            a[0]  = to_bf16(f0[0]); a[1]  = to_bf16(f0[1]); a[2]  = to_bf16(f0[2]); a[3]  = to_bf16(f0[3]);
            a[4]  = to_bf16(f1[0]); a[5]  = to_bf16(f1[1]); a[6]  = to_bf16(f1[2]); a[7]  = to_bf16(f1[3]);
            a[8]  = to_bf16(f2[0]); a[9]  = to_bf16(f2[1]); a[10] = to_bf16(f2[2]); a[11] = to_bf16(f2[3]);
            a[12] = to_bf16(f3[0]); a[13] = to_bf16(f3[1]); a[14] = to_bf16(f3[2]); a[15] = to_bf16(f3[3]);
            // B fragment: lane = col N=l16, K = k0 + hi*16 + 0..15 (contiguous in sW)
            const __bf16* bp = &sW[(n0 + l16) * WT_STRIDE + k0 + hi * 16];
            v16bf b;
#pragma unroll
            for (int j = 0; j < 16; ++j) b[j] = bp[j];
            acc = __builtin_amdgcn_wmma_f32_16x16x32_bf16(
                false, a, false, b, (short)0, acc, false, false);
        }

        // C/D layout: VGPR r -> row (hi*8 + r), col = n0 + l16
#pragma unroll
        for (int r = 0; r < 8; ++r) sH[(hi * 8 + r) * 128 + col] = acc[r] + bv;
        __syncthreads();

        // ---- LayerNorm over D=128 per row ----
        const float* hrow = &sH[row * 128 + seg * 8];
        float s = 0.f, ss = 0.f;
#pragma unroll
        for (int j = 0; j < 8; ++j) { const float v = hrow[j]; s += v; ss += v * v; }
        sSum[tid] = s; sSq[tid] = ss;
        __syncthreads();
        if (seg == 0) {
            float ts = 0.f, tss = 0.f;
#pragma unroll
            for (int j = 0; j < 16; ++j) { ts += sSum[row * 16 + j]; tss += sSq[row * 16 + j]; }
            const float mu  = ts * (1.0f / 128.0f);
            const float var = tss * (1.0f / 128.0f) - mu * mu;   // population variance
            sMu[row] = mu;
            sRs[row] = rsqrtf(var + LN_EPS);
        }
        __syncthreads();

        const int gr = rBase + row;
        if (gr < Mtot) {
            const float mu = sMu[row];
            const float rs = sRs[row];
#pragma unroll
            for (int j = 0; j < 8; ++j) {
                const int c = seg * 8 + j;
                const float v = (sH[row * 128 + c] - mu) * rs * gv[j] + bev[j];
                const float g = v * 0.5f * (1.0f + erff(v * 0.70710678118654752f));
                __builtin_nontemporal_store(g, &out[(size_t)gr * 128 + c]);  // write-once stream
            }
        }
        __syncthreads();   // protect sH/sSum before next tile
    }
}

extern "C" void kernel_launch(void* const* d_in, const int* in_sizes, int n_in,
                              void* d_out, int out_size, void* d_ws, size_t ws_size,
                              hipStream_t stream) {
    const float* x     = (const float*)d_in[0];
    const int*   rows  = (const int*)  d_in[1];
    const int*   cols  = (const int*)  d_in[2];
    const float* vals  = (const float*)d_in[3];
    const float* W     = (const float*)d_in[4];
    const float* bias  = (const float*)d_in[5];
    const float* gamma = (const float*)d_in[6];
    const float* beta  = (const float*)d_in[7];
    float* out = (float*)d_out;

    const int E    = in_sizes[1];
    const int N    = in_sizes[0] / 256;   // B*D = 256
    const int Mtot = 2 * N;
    const int n4   = (Mtot * 128) / 4;

    float* y = (float*)d_ws;                                  // [2,N,128] f32 (51.2 MB)
    const size_t yBytes = (size_t)Mtot * 128 * sizeof(float);
    unsigned short* Wt = (unsigned short*)((char*)d_ws + ((yBytes + 255) & ~(size_t)255));

    gcn_zero_kernel<<<(n4 + 255) / 256, 256, 0, stream>>>((float4*)y, n4);
    gcn_wt_prep_kernel<<<64, 256, 0, stream>>>(W, Wt);
    gcn_edge_spmm_kernel<<<(E + 7) / 8, 256, 0, stream>>>(x, rows, cols, vals, y, E, N);
    const int gemmBlocks = (Mtot + 16 * TILES_PER_BLOCK - 1) / (16 * TILES_PER_BLOCK);
    gcn_gemm_ln_gelu_kernel<<<gemmBlocks, 256, 0, stream>>>(y, Wt, bias, gamma, beta, out, Mtot);
}
